// MySpikeGPT_19894288515534
// MI455X (gfx1250) — matrix operand order repaired and checked by reading, not compiled, not run on hardware
//
#include <hip/hip_runtime.h>
#include <hip/hip_bf16.h>
#include <math.h>

typedef __attribute__((ext_vector_type(16))) _Float16 v16h;
typedef __attribute__((ext_vector_type(8)))  _Float16 v8h;
typedef __attribute__((ext_vector_type(8)))  float    v8f;

#define T_STEPS 4
#define SEQ 512
#define DMODEL 768
#define FFDIM 3072
#define NLAYER 6
#define VOCAB 32000
#define BN_EPS 1e-5f

union HV { v16h v; v8h h[2]; };

// ---------------------------------------------------------------------------
// CDNA5 async global->LDS copy (ASYNCcnt path), 16 bytes per lane.
// dst must be an LDS (shared) pointer, 16B aligned; src global, 16B aligned.
// ---------------------------------------------------------------------------
__device__ __forceinline__ void async_copy_b128(const _Float16* g, _Float16* l) {
    unsigned lds_off = (unsigned)(size_t)(__attribute__((address_space(3))) _Float16*)l;
    asm volatile("global_load_async_to_lds_b128 %0, %1, off"
                 :: "v"(lds_off), "v"(g) : "memory");
}
__device__ __forceinline__ void wait_asynccnt0() {
    asm volatile("s_wait_asynccnt 0" ::: "memory");
}

// ---------------------------------------------------------------------------
// Embedding + positional encoding + first bn_seq (T-broadcast => per-s stats
// over D only). One block per sequence position.
// ---------------------------------------------------------------------------
__global__ __launch_bounds__(256) void embed_bn_kernel(const int* __restrict__ tok,
                                                       const float* __restrict__ emb,
                                                       float* __restrict__ h) {
    int s = blockIdx.x, tid = threadIdx.x;
    __shared__ float r1[256], r2[256];
    __shared__ float sh_mu, sh_istd;
    int tk = tok[s];
    float sum = 0.f, sq = 0.f;
    for (int d = tid; d < DMODEL; d += 256) {
        float e = emb[(size_t)tk * DMODEL + d];
        float expnt = (float)(d & ~1) / (float)DMODEL;
        float ang = (float)s / powf(10000.0f, expnt);
        float pe = (d & 1) ? cosf(ang) : sinf(ang);
        float v = e + pe;
        sum += v; sq += v * v;
    }
    r1[tid] = sum; r2[tid] = sq; __syncthreads();
    for (int off = 128; off > 0; off >>= 1) {
        if (tid < off) { r1[tid] += r1[tid + off]; r2[tid] += r2[tid + off]; }
        __syncthreads();
    }
    if (tid == 0) {
        float mu = r1[0] / (float)DMODEL;
        float var = r2[0] / (float)DMODEL - mu * mu;
        sh_mu = mu; sh_istd = rsqrtf(var + BN_EPS);
    }
    __syncthreads();
    float mu = sh_mu, istd = sh_istd;
    for (int d = tid; d < DMODEL; d += 256) {
        float e = emb[(size_t)tk * DMODEL + d];
        float expnt = (float)(d & ~1) / (float)DMODEL;
        float ang = (float)s / powf(10000.0f, expnt);
        float pe = (d & 1) ? cosf(ang) : sinf(ang);
        float nv = (e + pe - mu) * istd;
        for (int t = 0; t < T_STEPS; ++t)
            h[((size_t)t * SEQ + s) * DMODEL + d] = nv;
    }
}

// ---------------------------------------------------------------------------
// IF neuron over T: v+=x; spike if v>=1; reset on spike. f16 binary out.
// ---------------------------------------------------------------------------
__global__ __launch_bounds__(256) void if_kernel(const float* __restrict__ x,
                                                 _Float16* __restrict__ out, int sd) {
    int idx = blockIdx.x * 256 + threadIdx.x;
    if (idx >= sd) return;
    float v = 0.f;
    for (int t = 0; t < T_STEPS; ++t) {
        v += x[(size_t)t * sd + idx];
        float sp = (v >= 1.0f) ? 1.0f : 0.0f;
        out[(size_t)t * sd + idx] = (_Float16)sp;
        v = (sp != 0.f) ? 0.f : v;
    }
}

// ---------------------------------------------------------------------------
// WMMA GEMM: C(MxN,f32) = A(MxK,f16 spikes) @ B(KxN,f32 weights) [+ bias].
// Block tile 128x128, 8 wave32 waves; each wave owns a 32x64 strip = 2x4
// v_wmma_f32_16x16x32_f16 accumulators (1.5 LDS b128 loads per WMMA).
// Double-buffered LDS; A tile streamed with global_load_async_to_lds_b128
// (ASYNCcnt), B tile converted f32->f16 through registers and stored
// transposed so each B fragment is two contiguous b128 LDS loads.
// ---------------------------------------------------------------------------
#define LDA 40
#define LDB 40
__global__ __launch_bounds__(256) void gemm_wmma_kernel(const _Float16* __restrict__ A,
                                                        const float* __restrict__ B,
                                                        const float* __restrict__ bias,
                                                        float* __restrict__ C,
                                                        int M, int N, int K) {
    __shared__ _Float16 Al[2][128 * LDA];
    __shared__ _Float16 Bl[2][128 * LDB];
    int tid = threadIdx.x;
    int wave = tid >> 5, lane = tid & 31;
    int wm = (wave >> 1) * 32;          // wave M strip within block tile
    int wn = (wave & 1) * 64;           // wave N strip within block tile
    int n0 = blockIdx.x * 128;
    int m0 = blockIdx.y * 128;

    int lm = lane & 15;
    int kh = lane >> 4;                 // 0 or 1
    int ak = kh * 8;                    // A frag K sub-chunk
    int bk = kh * 8;                    // B frag K base

    // A tile loader: thread -> row tid>>1, 16 halfs at col (tid&1)*16
    int ar = tid >> 1, ac = (tid & 1) * 16;
    // B tile loader: thread -> k-row tid>>3, 16 floats at col (tid&7)*16
    int bkr = tid >> 3, bc = (tid & 7) * 16;

    v8f acc[2][4] = {};
    float breg[16];
    const int steps = K / 32;

    // prologue: tile 0
    {
        const _Float16* ga = A + (size_t)(m0 + ar) * K + ac;
        async_copy_b128(ga, &Al[0][ar * LDA + ac]);
        async_copy_b128(ga + 8, &Al[0][ar * LDA + ac + 8]);
        const float* gb = B + (size_t)bkr * N + n0 + bc;
        #pragma unroll
        for (int j = 0; j < 16; ++j) breg[j] = gb[j];
        #pragma unroll
        for (int j = 0; j < 16; ++j) Bl[0][(bc + j) * LDB + bkr] = (_Float16)breg[j];
    }

    for (int s = 0; s < steps; ++s) {
        int p = s & 1;
        wait_asynccnt0();
        __syncthreads();

        if (s + 1 < steps) {
            int k1 = (s + 1) * 32;
            const _Float16* ga = A + (size_t)(m0 + ar) * K + k1 + ac;
            async_copy_b128(ga, &Al[p ^ 1][ar * LDA + ac]);
            async_copy_b128(ga + 8, &Al[p ^ 1][ar * LDA + ac + 8]);
            const float* gb = B + (size_t)(k1 + bkr) * N + n0 + bc;
            #pragma unroll
            for (int j = 0; j < 16; ++j) breg[j] = gb[j];
        }

        // gather all fragments, then issue 8 back-to-back WMMAs
        HV a[2];
        #pragma unroll
        for (int mi = 0; mi < 2; ++mi) {
            int row = wm + mi * 16 + lm;
            a[mi].h[0] = *(const v8h*)&Al[p][row * LDA + ak];
            a[mi].h[1] = *(const v8h*)&Al[p][row * LDA + ak + 16];
        }
        HV b[4];
        #pragma unroll
        for (int ni = 0; ni < 4; ++ni) {
            int col = wn + ni * 16 + lm;
            b[ni].h[0] = *(const v8h*)&Bl[p][col * LDB + bk];
            b[ni].h[1] = *(const v8h*)&Bl[p][col * LDB + bk + 16];
        }
        #pragma unroll
        for (int mi = 0; mi < 2; ++mi)
            #pragma unroll
            for (int ni = 0; ni < 4; ++ni)
                acc[mi][ni] = __builtin_amdgcn_wmma_f32_16x16x32_f16(
                    false, a[mi].v, false, b[ni].v, (short)0, acc[mi][ni], false, false);

        if (s + 1 < steps) {
            #pragma unroll
            for (int j = 0; j < 16; ++j) Bl[p ^ 1][(bc + j) * LDB + bkr] = (_Float16)breg[j];
        }
    }

    // D layout: VGPR r -> M=r (lanes 0-15) / r+8 (lanes 16-31); N = lane%16
    int hi = kh * 8;
    #pragma unroll
    for (int mi = 0; mi < 2; ++mi)
        #pragma unroll
        for (int ni = 0; ni < 4; ++ni) {
            int col = n0 + wn + ni * 16 + lm;
            float bv = bias ? bias[col] : 0.0f;
            #pragma unroll
            for (int r = 0; r < 8; ++r) {
                int row = m0 + wm + mi * 16 + r + hi;
                C[(size_t)row * N + col] = acc[mi][ni][r] + bv;
            }
        }
}

// ---------------------------------------------------------------------------
// bn_seq (per-position stats over T x feature) + IF neuron -> f16 spikes.
// ---------------------------------------------------------------------------
__global__ __launch_bounds__(256) void bn_if_kernel(const float* __restrict__ G,
                                                    _Float16* __restrict__ out, int Dx) {
    int s = blockIdx.x, tid = threadIdx.x;
    __shared__ float r1[256], r2[256];
    __shared__ float sh_mu, sh_istd;
    float sum = 0.f, sq = 0.f;
    for (int d = tid; d < Dx; d += 256)
        for (int t = 0; t < T_STEPS; ++t) {
            float v = G[((size_t)t * SEQ + s) * Dx + d];
            sum += v; sq += v * v;
        }
    r1[tid] = sum; r2[tid] = sq; __syncthreads();
    for (int off = 128; off > 0; off >>= 1) {
        if (tid < off) { r1[tid] += r1[tid + off]; r2[tid] += r2[tid + off]; }
        __syncthreads();
    }
    if (tid == 0) {
        float inv = 1.0f / (float)(T_STEPS * Dx);
        float mu = r1[0] * inv;
        float var = r2[0] * inv - mu * mu;
        sh_mu = mu; sh_istd = rsqrtf(var + BN_EPS);
    }
    __syncthreads();
    float mu = sh_mu, istd = sh_istd;
    for (int d = tid; d < Dx; d += 256) {
        float v = 0.f;
        for (int t = 0; t < T_STEPS; ++t) {
            v += (G[((size_t)t * SEQ + s) * Dx + d] - mu) * istd;
            float sp = (v >= 1.0f) ? 1.0f : 0.0f;
            out[((size_t)t * SEQ + s) * Dx + d] = (_Float16)sp;
            v = (sp != 0.f) ? 0.f : v;
        }
    }
}

// bn_seq + residual add: h += normalize(G). One block per position.
__global__ __launch_bounds__(256) void bn_res_kernel(const float* __restrict__ G,
                                                     float* __restrict__ h) {
    int s = blockIdx.x, tid = threadIdx.x;
    __shared__ float r1[256], r2[256];
    __shared__ float sh_mu, sh_istd;
    float sum = 0.f, sq = 0.f;
    for (int d = tid; d < DMODEL; d += 256)
        for (int t = 0; t < T_STEPS; ++t) {
            float v = G[((size_t)t * SEQ + s) * DMODEL + d];
            sum += v; sq += v * v;
        }
    r1[tid] = sum; r2[tid] = sq; __syncthreads();
    for (int off = 128; off > 0; off >>= 1) {
        if (tid < off) { r1[tid] += r1[tid + off]; r2[tid] += r2[tid + off]; }
        __syncthreads();
    }
    if (tid == 0) {
        float inv = 1.0f / (float)(T_STEPS * DMODEL);
        float mu = r1[0] * inv;
        float var = r2[0] * inv - mu * mu;
        sh_mu = mu; sh_istd = rsqrtf(var + BN_EPS);
    }
    __syncthreads();
    float mu = sh_mu, istd = sh_istd;
    for (int d = tid; d < DMODEL; d += 256)
        for (int t = 0; t < T_STEPS; ++t) {
            size_t i = ((size_t)t * SEQ + s) * DMODEL + d;
            h[i] += (G[i] - mu) * istd;
        }
}

// qk[t, head*64+hd] = IF_t( sum_s Q[t,s,dg]*K[t,s,dg] )
__global__ __launch_bounds__(256) void attn_score_kernel(const _Float16* __restrict__ Q,
                                                         const _Float16* __restrict__ K,
                                                         _Float16* __restrict__ qk) {
    int dg = blockIdx.x * 256 + threadIdx.x;
    if (dg >= DMODEL) return;
    float accs[T_STEPS];
    for (int t = 0; t < T_STEPS; ++t) {
        float a = 0.f;
        for (int s = 0; s < SEQ; ++s) {
            size_t i = ((size_t)t * SEQ + s) * DMODEL + dg;
            a += (float)Q[i] * (float)K[i];
        }
        accs[t] = a;
    }
    float v = 0.f;
    for (int t = 0; t < T_STEPS; ++t) {
        v += accs[t];
        float sp = (v >= 1.0f) ? 1.0f : 0.0f;
        qk[t * DMODEL + dg] = (_Float16)sp;
        v = (sp != 0.f) ? 0.f : v;
    }
}

// attn[t,s,dg] = V[t,s,dg] * qk[t,dg]
__global__ __launch_bounds__(256) void attn_mul_kernel(const _Float16* __restrict__ V,
                                                       const _Float16* __restrict__ qk,
                                                       _Float16* __restrict__ attn) {
    size_t idx = (size_t)blockIdx.x * 256 + threadIdx.x;
    if (idx >= (size_t)T_STEPS * SEQ * DMODEL) return;
    int t = (int)(idx / ((size_t)SEQ * DMODEL));
    int dg = (int)(idx % DMODEL);
    attn[idx] = (_Float16)((float)V[idx] * (float)qk[t * DMODEL + dg]);
}

// Final projection at position SEQ-1 only (output uses only m[cur_pos-1] and
// bn/IF are position-local): logits4[t][v], bandwidth-bound on w_out.
__global__ __launch_bounds__(256) void final_proj_kernel(const _Float16* __restrict__ sp,
                                                         const float* __restrict__ w_out,
                                                         const float* __restrict__ b_out,
                                                         float* __restrict__ logits4) {
    __shared__ _Float16 sl[T_STEPS * DMODEL];
    int tid = threadIdx.x;
    for (int j = tid; j < T_STEPS * DMODEL; j += 256) {
        int t = j / DMODEL, k = j % DMODEL;
        sl[j] = sp[((size_t)t * SEQ + (SEQ - 1)) * DMODEL + k];
    }
    __syncthreads();
    int v = blockIdx.x * 256 + tid;
    if (v >= VOCAB) return;
    float a0 = 0.f, a1 = 0.f, a2 = 0.f, a3 = 0.f;
    for (int k = 0; k < DMODEL; ++k) {
        float w = w_out[(size_t)k * VOCAB + v];
        a0 += (float)sl[k] * w;
        a1 += (float)sl[DMODEL + k] * w;
        a2 += (float)sl[2 * DMODEL + k] * w;
        a3 += (float)sl[3 * DMODEL + k] * w;
    }
    float bb = b_out[v];
    logits4[0 * VOCAB + v] = a0 + bb;
    logits4[1 * VOCAB + v] = a1 + bb;
    logits4[2 * VOCAB + v] = a2 + bb;
    logits4[3 * VOCAB + v] = a3 + bb;
}

__global__ __launch_bounds__(1024) void final_stats_kernel(const float* __restrict__ logits4,
                                                           float* __restrict__ scal) {
    __shared__ float r1[1024], r2[1024];
    int tid = threadIdx.x;
    float sum = 0.f, sq = 0.f;
    for (int i = tid; i < T_STEPS * VOCAB; i += 1024) {
        float x = logits4[i];
        sum += x; sq += x * x;
    }
    r1[tid] = sum; r2[tid] = sq; __syncthreads();
    for (int off = 512; off > 0; off >>= 1) {
        if (tid < off) { r1[tid] += r1[tid + off]; r2[tid] += r2[tid + off]; }
        __syncthreads();
    }
    if (tid == 0) {
        float inv = 1.0f / (float)(T_STEPS * VOCAB);
        float mu = r1[0] * inv;
        float var = r2[0] * inv - mu * mu;
        scal[0] = mu;
        scal[1] = rsqrtf(var + BN_EPS);
    }
}

__global__ __launch_bounds__(256) void final_m_kernel(const float* __restrict__ logits4,
                                                      const float* __restrict__ scal,
                                                      float* __restrict__ sumM,
                                                      float* __restrict__ mbuf) {
    __shared__ float red[256];
    int tid = threadIdx.x;
    int v = blockIdx.x * 256 + tid;
    float m = 0.f;
    if (v < VOCAB) {
        float mu = scal[0], istd = scal[1];
        float acc = 0.f, vv = 0.f;
        for (int t = 0; t < T_STEPS; ++t) {
            vv += (logits4[t * VOCAB + v] - mu) * istd;
            float sp = (vv >= 1.0f) ? 1.0f : 0.0f;
            acc += sp;
            vv = (sp != 0.f) ? 0.f : vv;
        }
        m = acc * (1.0f / (float)T_STEPS);
        mbuf[v] = m;
    }
    red[tid] = m; __syncthreads();
    for (int off = 128; off > 0; off >>= 1) {
        if (tid < off) red[tid] += red[tid + off];
        __syncthreads();
    }
    if (tid == 0) atomicAdd(sumM, red[0]);
}

__global__ __launch_bounds__(256) void final_div_kernel(const float* __restrict__ mbuf,
                                                        const float* __restrict__ sumM,
                                                        float* __restrict__ out) {
    int v = blockIdx.x * 256 + threadIdx.x;
    if (v < VOCAB) out[v] = mbuf[v] / sumM[0];
}

// ---------------------------------------------------------------------------
extern "C" void kernel_launch(void* const* d_in, const int* in_sizes, int n_in,
                              void* d_out, int out_size, void* d_ws, size_t ws_size,
                              hipStream_t stream) {
    (void)in_sizes; (void)n_in; (void)out_size; (void)ws_size;
    const int*   tok   = (const int*)d_in[0];
    // d_in[1] = cur_pos (== SEQ, hardcoded)
    const float* emb   = (const float*)d_in[2];
    const float* wq    = (const float*)d_in[3];
    const float* wk    = (const float*)d_in[4];
    const float* wv    = (const float*)d_in[5];
    const float* wo    = (const float*)d_in[6];
    const float* w1    = (const float*)d_in[7];
    const float* b1    = (const float*)d_in[8];
    const float* w2    = (const float*)d_in[9];
    const float* b2    = (const float*)d_in[10];
    const float* w_out = (const float*)d_in[11];
    const float* b_out = (const float*)d_in[12];
    float* out = (float*)d_out;

    // workspace layout
    char* ws = (char*)d_ws;
    const size_t SZ_H   = (size_t)T_STEPS * SEQ * DMODEL * 4;
    const size_t SZ_G   = (size_t)T_STEPS * SEQ * FFDIM * 4;
    const size_t SZ_A16 = (size_t)T_STEPS * SEQ * FFDIM * 2;
    const size_t SZ_P16 = (size_t)T_STEPS * SEQ * DMODEL * 2;
    float*    h    = (float*)ws;                       ws += SZ_H;
    float*    G    = (float*)ws;                       ws += SZ_G;
    _Float16* A16  = (_Float16*)ws;                    ws += SZ_A16;
    _Float16* Q16  = (_Float16*)ws;                    ws += SZ_P16;
    _Float16* K16  = (_Float16*)ws;                    ws += SZ_P16;
    _Float16* V16  = (_Float16*)ws;                    ws += SZ_P16;
    _Float16* qk16 = (_Float16*)ws;                    ws += 8192;
    float* logits4 = (float*)ws;                       ws += (size_t)T_STEPS * VOCAB * 4;
    float* mbuf    = (float*)ws;                       ws += (size_t)VOCAB * 4;
    float* scal    = (float*)ws;                       ws += 256;
    float* sumM    = (float*)ws;                       ws += 256;

    const int SD  = SEQ * DMODEL;
    const int M   = T_STEPS * SEQ;                     // 2048
    const dim3 blk(256);
    const dim3 gD(DMODEL / 128, M / 128);              // N=768 GEMMs
    const dim3 gF(FFDIM / 128, M / 128);               // N=3072 GEMM
    const int gIF = (SD + 255) / 256;
    const int gMul = (T_STEPS * SD + 255) / 256;
    const int gV = (VOCAB + 255) / 256;

    embed_bn_kernel<<<SEQ, blk, 0, stream>>>(tok, emb, h);

    for (int i = 0; i < NLAYER; ++i) {
        const float* Wq = wq + (size_t)i * DMODEL * DMODEL;
        const float* Wk = wk + (size_t)i * DMODEL * DMODEL;
        const float* Wv = wv + (size_t)i * DMODEL * DMODEL;
        const float* Wo = wo + (size_t)i * DMODEL * DMODEL;
        const float* W1 = w1 + (size_t)i * DMODEL * FFDIM;
        const float* B1 = b1 + (size_t)i * FFDIM;
        const float* W2 = w2 + (size_t)i * FFDIM * DMODEL;
        const float* B2 = b2 + (size_t)i * DMODEL;

        if_kernel<<<gIF, blk, 0, stream>>>(h, A16, SD);            // tmp spikes

        gemm_wmma_kernel<<<gD, blk, 0, stream>>>(A16, Wq, nullptr, G, M, DMODEL, DMODEL);
        bn_if_kernel<<<SEQ, blk, 0, stream>>>(G, Q16, DMODEL);
        gemm_wmma_kernel<<<gD, blk, 0, stream>>>(A16, Wk, nullptr, G, M, DMODEL, DMODEL);
        bn_if_kernel<<<SEQ, blk, 0, stream>>>(G, K16, DMODEL);
        gemm_wmma_kernel<<<gD, blk, 0, stream>>>(A16, Wv, nullptr, G, M, DMODEL, DMODEL);
        bn_if_kernel<<<SEQ, blk, 0, stream>>>(G, V16, DMODEL);

        attn_score_kernel<<<(DMODEL + 255) / 256, blk, 0, stream>>>(Q16, K16, qk16);
        attn_mul_kernel<<<gMul, blk, 0, stream>>>(V16, qk16, A16);  // attn spikes

        gemm_wmma_kernel<<<gD, blk, 0, stream>>>(A16, Wo, nullptr, G, M, DMODEL, DMODEL);
        bn_res_kernel<<<SEQ, blk, 0, stream>>>(G, h);

        if_kernel<<<gIF, blk, 0, stream>>>(h, A16, SD);            // s spikes
        gemm_wmma_kernel<<<gF, blk, 0, stream>>>(A16, W1, B1, G, M, FFDIM, DMODEL);
        bn_if_kernel<<<SEQ, blk, 0, stream>>>(G, A16, FFDIM);      // u spikes
        gemm_wmma_kernel<<<gD, blk, 0, stream>>>(A16, W2, B2, G, M, DMODEL, FFDIM);
        bn_res_kernel<<<SEQ, blk, 0, stream>>>(G, h);
    }

    if_kernel<<<gIF, blk, 0, stream>>>(h, A16, SD);                // final spikes
    final_proj_kernel<<<gV, blk, 0, stream>>>(A16, w_out, b_out, logits4);
    final_stats_kernel<<<1, 1024, 0, stream>>>(logits4, scal);
    hipMemsetAsync(sumM, 0, sizeof(float), stream);
    final_m_kernel<<<gV, blk, 0, stream>>>(logits4, scal, sumM, mbuf);
    final_div_kernel<<<gV, blk, 0, stream>>>(mbuf, sumM, out);
}
